// CAAM_16346645528937
// MI455X (gfx1250) — matrix operand order repaired and checked by reading, not compiled, not run on hardware
//
#include <hip/hip_runtime.h>
#include <hip/hip_bf16.h>
#include <math.h>
#include <stdint.h>

// ---------------------------------------------------------------------------
// Problem constants
// ---------------------------------------------------------------------------
#define Bb   8
#define Cc   512
#define Hh   128
#define Ww   128
#define HW   16384     // H*W
#define Kk   19
#define Nn   16        // BH*BW
#define Pp   1024      // rH*rW
#define CI   256

typedef __attribute__((ext_vector_type(16))) _Float16 v16h;
typedef __attribute__((ext_vector_type(8)))  _Float16 v8h;
typedef __attribute__((ext_vector_type(4)))  _Float16 v4h;
typedef __attribute__((ext_vector_type(8)))  float    v8f;
typedef __attribute__((ext_vector_type(4)))  float    f32x4;
typedef __attribute__((ext_vector_type(4)))  unsigned int v4u;
typedef __attribute__((ext_vector_type(8)))  int      v8i;
typedef __attribute__((ext_vector_type(4)))  int      v4i;

#if defined(__has_builtin)
#if __has_builtin(__builtin_amdgcn_tensor_load_to_lds)
#define HAVE_TDM 1
#endif
#endif

// ---------------------------------------------------------------------------
// WMMA helpers (CDNA5 wave32 16x16x32 f16 -> f32)
//
// A fragment (16x32 MxK, f16, row-major in LDS, stride lda halves):
//   lanes 0-15 : row M=lane,    halves K0..15
//   lanes16-31 : row M=lane-16, halves K16..31
// B tile (32x16 KxN): fragment-native LDS layout [col][k]:
//   lanes 0-15 : col N=lane,    halves K0..15
//   lanes16-31 : col N=lane-16, halves K16..31
// Both -> two contiguous ds_load_b128 per lane.
// C/D: v8f, lane<16: N=lane, M=i ; lane>=16: N=lane-16, M=8+i
// ---------------------------------------------------------------------------
#define BTILE 512           // halves per fragment-native B tile (16 cols * 32 k)
#define BIDX(d, cc) ((((d) >> 4) * BTILE) + (((d) & 15) * 32) + (cc))

static __device__ inline v8f wmma_f16(v16h a, v16h b, v8f c) {
  return __builtin_amdgcn_wmma_f32_16x16x32_f16(
      false, a, false, b, (short)0, c, false, false);
}

static __device__ inline v16h combine16(v8h lo, v8h hi) {
  return __builtin_shufflevector(lo, hi, 0, 1, 2, 3, 4, 5, 6, 7,
                                 8, 9, 10, 11, 12, 13, 14, 15);
}

static __device__ inline v16h load_a_frag(const _Float16* p, int lda, int lane) {
  const int row  = lane & 15;
  const int koff = (lane >> 4) * 16;
  const _Float16* q = p + row * lda + koff;
  v8h lo = *(const v8h*)(q);        // K koff..koff+7
  v8h hi = *(const v8h*)(q + 8);    // K koff+8..koff+15
  return combine16(lo, hi);
}

static __device__ inline v16h load_b_frag(const _Float16* p, int lane) {
  const int col = lane & 15;
  const int kb  = (lane >> 4) * 16;
  const _Float16* q = p + col * 32 + kb;
  v8h lo = *(const v8h*)(q);
  v8h hi = *(const v8h*)(q + 8);
  return combine16(lo, hi);
}

static __device__ inline v4h cvt4(f32x4 v) {
  v4h r;
  r[0] = (_Float16)v[0];
  r[1] = (_Float16)v[1];
  r[2] = (_Float16)v[2];
  r[3] = (_Float16)v[3];
  return r;
}

static __device__ inline float prelu_f(float v, float a) {
  return (v >= 0.f) ? v : a * v;
}

// ---------------------------------------------------------------------------
// Tensor Data Mover: 2D f32 tile -> LDS (D# per ISA 08_async_tensor §8)
// ---------------------------------------------------------------------------
#ifdef HAVE_TDM
static __device__ inline void tdm_load_tile_f32(
    unsigned int lds_byte_addr, const float* gaddr,
    int tensor_d0, int tensor_d1, int tile_d0, int tile_d1, int stride0) {
  unsigned long long ga = (unsigned long long)(uintptr_t)gaddr;
  v4u g0;
  g0[0] = 1u;                                        // count=1, user desc
  g0[1] = lds_byte_addr;                             // lds_addr [63:32]
  g0[2] = (unsigned int)(ga & 0xffffffffu);          // global_addr lo
  g0[3] = (unsigned int)((ga >> 32) & 0x01ffffffu)   // global_addr hi
          | 0x80000000u;                             // type=2 ("image")
  v8i g1;
  g1[0] = (2 << 16);                                 // data_size=4B, mask=0
  g1[1] = (int)((unsigned)(tensor_d0 & 0xffff) << 16);         // dim0 lo16
  g1[2] = (int)(((unsigned)tensor_d0 >> 16) & 0xffffu)         // dim0 hi16
          | (int)((unsigned)(tensor_d1 & 0xffff) << 16);       // dim1 lo16
  g1[3] = (int)(((unsigned)tensor_d1 >> 16) & 0xffffu)         // dim1 hi16
          | (int)((unsigned)(tile_d0 & 0xffff) << 16);         // tile_dim0
  g1[4] = (tile_d1 & 0xffff);                        // tile_dim1 (tile_dim2=0)
  g1[5] = stride0;                                   // dim0_stride [191:160]
  g1[6] = 0;
  g1[7] = 0;
  v4i z = {};
#if defined(__clang_major__) && __clang_major__ >= 23
  v8i z8 = {};
  __builtin_amdgcn_tensor_load_to_lds(g0, g1, z, z, z8, 0);
#else
  __builtin_amdgcn_tensor_load_to_lds(g0, g1, z, z, 0);
#endif
}
#endif

// ---------------------------------------------------------------------------
// Kernel 1: cam[b,k,h,w] = conv_cam_w[k,:] . x[b,:,h,w] + bias
// grid (HW/256, B), block 256 (8 waves). M=19 (pad 32), N=256/blk, K=512.
// ---------------------------------------------------------------------------
__global__ __launch_bounds__(256) void cam_kernel(
    const float* __restrict__ x, const float* __restrict__ w,
    const float* __restrict__ bias, float* __restrict__ cam) {
  __shared__ _Float16 Wl[32 * 512];   // A: 19(pad32) x 512, row-major
  __shared__ _Float16 Xl[16 * BTILE]; // B chunk: 32ch x 256px, tile-native
  const int b  = blockIdx.y;
  const int p0 = blockIdx.x * 256;
  const int t  = threadIdx.x, wv = t >> 5, lane = t & 31;
  const int mt = wv & 1;              // wave-constant M tile

  for (int e = t; e < 32 * 128; e += 256) {     // float4-vectorized W stage
    int k = e >> 7, c4 = (e & 127) * 4;
    v4h val = {};
    if (k < Kk) val = cvt4(*(const f32x4*)(w + k * Cc + c4));
    *(v4h*)(Wl + k * 512 + c4) = val;
  }
  __syncthreads();

  v8f acc[4] = {};
  for (int kc = 0; kc < 16; ++kc) {
    __builtin_prefetch(x + ((size_t)(b * Cc + ((kc + 1) & 15) * 32)) * HW + p0, 0, 0);
    for (int e = t; e < 32 * 64; e += 256) {    // float4 over pixels
      int row = e >> 6, col4 = (e & 63) * 4;    // row = ch in chunk
      f32x4 vv = *(const f32x4*)(x + ((size_t)(b * Cc + kc * 32 + row)) * HW + p0 + col4);
#pragma unroll
      for (int i = 0; i < 4; ++i)
        Xl[BIDX(col4 + i, row)] = (_Float16)vv[i];
    }
    __syncthreads();
    v16h af = load_a_frag(Wl + mt * 16 * 512 + kc * 32, 512, lane);
#pragma unroll
    for (int j = 0; j < 4; ++j) {
      int nt = (wv + j * 8) >> 1;
      v16h bf = load_b_frag(Xl + nt * BTILE, lane);
      acc[j] = wmma_f16(af, bf, acc[j]);
    }
    __syncthreads();
  }
  const int nn = lane & 15, mb = (lane >> 4) * 8;
#pragma unroll
  for (int j = 0; j < 4; ++j) {
    int nt = (wv + j * 8) >> 1;
#pragma unroll
    for (int i = 0; i < 8; ++i) {
      int kg = mt * 16 + mb + i;
      if (kg < Kk)
        cam[((size_t)(b * Kk + kg)) * HW + p0 + nt * 16 + nn] = acc[j][i] + bias[kg];
    }
  }
}

// ---------------------------------------------------------------------------
// Kernel 2 (fused pool + pixel softmax), bin tile via Tensor Data Mover.
// grid (16,19,8), block 256
// ---------------------------------------------------------------------------
__global__ __launch_bounds__(256) void binstat_kernel(
    const float* __restrict__ cam, float* __restrict__ cls,
    float* __restrict__ conf) {
  __shared__ float tile[Pp];          // 32x32 bin tile
  __shared__ float red[256];
  const int n = blockIdx.x, k = blockIdx.y, b = blockIdx.z;
  const int bh = n >> 2, bw = n & 3;
  const float* gsrc =
      cam + ((size_t)(b * Kk + k)) * HW + (bh * 32) * Ww + bw * 32;
  const int t = threadIdx.x;

#ifdef HAVE_TDM
  if (t < 32) {                       // wave 0 issues the DMA once
    tdm_load_tile_f32((unsigned int)(uintptr_t)&tile[0], gsrc,
                      Ww, Hh, 32, 32, Ww);
    __builtin_amdgcn_s_wait_tensorcnt(0);
  }
  __syncthreads();
#else
  for (int j = t; j < Pp; j += 256) tile[j] = gsrc[(j >> 5) * Ww + (j & 31)];
  __syncthreads();
#endif

  float s = 0.f, mx = -3.4e38f;
  for (int j = t; j < Pp; j += 256) {
    float v = tile[j];
    s += v;
    mx = fmaxf(mx, v);
  }
  red[t] = s;
  __syncthreads();
  for (int st = 128; st > 0; st >>= 1) {
    if (t < st) red[t] += red[t + st];
    __syncthreads();
  }
  if (t == 0) {
    float m = red[0] * (1.f / 1024.f);
    cls[((size_t)(b * Kk + k)) * Nn + n] = 1.f / (1.f + expf(-m));
  }
  __syncthreads();
  red[t] = mx;
  __syncthreads();
  for (int st = 128; st > 0; st >>= 1) {
    if (t < st) red[t] = fmaxf(red[t], red[t + st]);
    __syncthreads();
  }
  mx = red[0];
  __syncthreads();
  float es = 0.f;
  for (int j = t; j < Pp; j += 256) es += expf(tile[j] - mx);
  red[t] = es;
  __syncthreads();
  for (int st = 128; st > 0; st >>= 1) {
    if (t < st) red[t] += red[t + st];
    __syncthreads();
  }
  const float inv = 1.f / red[0];
  float* outp = conf + (((size_t)(b * Nn + n)) * Kk + k) * Pp;
  for (int j = t; j < Pp; j += 256) outp[j] = expf(tile[j] - mx) * inv;
}

// ---------------------------------------------------------------------------
// Kernel 3: local[b,n,k,c] = (sum_p conf[b,n,k,p] * x_p[b,n,p,c]) * bin_conf
// one block per (b,n). M=19(pad32), N=512, Kdim=1024
// ---------------------------------------------------------------------------
__global__ __launch_bounds__(256) void local_kernel(
    const float* __restrict__ x, const float* __restrict__ conf,
    const float* __restrict__ cls, float* __restrict__ loc) {
  __shared__ _Float16 Al[32 * 32];     // conf chunk: k x p, row-major
  __shared__ _Float16 Bl[32 * BTILE];  // x chunk: 32px x 512ch, tile-native
  const int bn = blockIdx.x;
  const int b = bn >> 4, n = bn & 15;
  const int bh = n >> 2, bw = n & 3;
  const int t = threadIdx.x, wv = t >> 5, lane = t & 31;
  const int mt = wv & 1;

  v8f acc[8] = {};
  for (int pc = 0; pc < 32; ++pc) {    // 32 K-chunks of 32 px (one rh row each)
    for (int e = t; e < 32 * 8; e += 256) {       // float4 conf stage
      int kr = e >> 3, pp4 = (e & 7) * 4;
      v4h val = {};
      if (kr < Kk)
        val = cvt4(*(const f32x4*)(conf + (((size_t)bn) * Kk + kr) * Pp + pc * 32 + pp4));
      *(v4h*)(Al + kr * 32 + pp4) = val;
    }
    for (int e = t; e < 512 * 8; e += 256) {      // float4 over pixels
      int c = e >> 3, pr4 = (e & 7) * 4;
      int hw = (bh * 32 + pc) * Ww + bw * 32 + pr4;
      f32x4 vv = *(const f32x4*)(x + ((size_t)(b * Cc + c)) * HW + hw);
      *(v4h*)(Bl + BIDX(c, pr4)) = cvt4(vv);      // k-contiguous -> b64 store
    }
    __syncthreads();
    v16h af = load_a_frag(Al + mt * 16 * 32, 32, lane);
#pragma unroll
    for (int j = 0; j < 8; ++j) {
      int nt = (wv + j * 8) >> 1;      // 0..31
      v16h bf = load_b_frag(Bl + nt * BTILE, lane);
      acc[j] = wmma_f16(af, bf, acc[j]);
    }
    __syncthreads();
  }
  const int nn = lane & 15, mb = (lane >> 4) * 8;
#pragma unroll
  for (int j = 0; j < 8; ++j) {
    int nt = (wv + j * 8) >> 1;
#pragma unroll
    for (int i = 0; i < 8; ++i) {
      int kg = mt * 16 + mb + i;
      if (kg < Kk) {
        float bc = cls[((size_t)(b * Kk + kg)) * Nn + n];
        loc[(((size_t)bn) * Kk + kg) * Cc + nt * 16 + nn] = acc[j][i] * bc;
      }
    }
  }
}

// ---------------------------------------------------------------------------
// Kernel 4: GCN 16x16 node mix + residual + PReLU (one thread per (b,k,c))
// ---------------------------------------------------------------------------
__global__ __launch_bounds__(256) void gcn_kernel(
    const float* __restrict__ loc, const float* __restrict__ w1,
    const float* __restrict__ pa, float* __restrict__ g) {
  const int idx = blockIdx.x * 256 + threadIdx.x;
  if (idx >= Bb * Kk * Cc) return;
  const int c = idx & 511, bk = idx >> 9;
  const int k = bk % Kk, b = bk / Kk;
  float lv[16];
#pragma unroll
  for (int n = 0; n < 16; ++n)
    lv[n] = loc[(((size_t)(b * Nn + n)) * Kk + k) * Cc + c];
#pragma unroll
  for (int m = 0; m < 16; ++m) {
    float s = 0.f;
#pragma unroll
    for (int n = 0; n < 16; ++n) s += w1[m * 16 + n] * lv[n];
    s += lv[m];
    g[(((size_t)(b * Nn + m)) * Kk + k) * Cc + c] = prelu_f(s, pa[m]);
  }
}

// ---------------------------------------------------------------------------
// Kernel 5/6: generic row GEMM: out[r,d] = sum_c A[r,c]*W[d,c] (+bias[d])
// rows = 2432 (152 M-tiles), Kdim = 512. grid (Ncols/128, 152), block 256.
// ---------------------------------------------------------------------------
__global__ __launch_bounds__(256) void rowgemm_kernel(
    const float* __restrict__ Ain, const float* __restrict__ W,
    const float* __restrict__ bias, float* __restrict__ out, int Ncols) {
  __shared__ _Float16 Al[16 * 32];
  __shared__ _Float16 Bl[8 * BTILE];   // 128 cols x 32 k, tile-native
  const int mtile = blockIdx.y;
  const int nbase = blockIdx.x * 128;
  const int t = threadIdx.x, wv = t >> 5, lane = t & 31;

  v8f acc = {};
  for (int kc = 0; kc < 16; ++kc) {
    __builtin_prefetch(W + ((size_t)nbase) * Cc + ((kc + 1) & 15) * 32, 0, 0);
    if (t < 16 * 8) {                             // float4 A stage
      int r = t >> 3, c4 = (t & 7) * 4;
      f32x4 vv = *(const f32x4*)(Ain + ((size_t)(mtile * 16 + r)) * Cc + kc * 32 + c4);
      *(v4h*)(Al + r * 32 + c4) = cvt4(vv);
    }
    for (int e = t; e < 128 * 8; e += 256) {      // float4 B stage
      int d = e >> 3, cc4 = (e & 7) * 4;
      f32x4 vv = *(const f32x4*)(W + ((size_t)(nbase + d)) * Cc + kc * 32 + cc4);
      *(v4h*)(Bl + BIDX(d, cc4)) = cvt4(vv);
    }
    __syncthreads();
    v16h af = load_a_frag(Al, 32, lane);
    v16h bf = load_b_frag(Bl + wv * BTILE, lane);
    acc = wmma_f16(af, bf, acc);
    __syncthreads();
  }
  const int nn = lane & 15, mb = (lane >> 4) * 8;
  const int d = nbase + wv * 16 + nn;
  const float bd = bias ? bias[d] : 0.f;
#pragma unroll
  for (int i = 0; i < 8; ++i) {
    int r = mtile * 16 + mb + i;
    out[(size_t)r * Ncols + d] = acc[i] + bd;
  }
}

// ---------------------------------------------------------------------------
// Kernel 7: glob[b,k,c] = prelu(sum_n fuse_w[n]*local_g + fuse_b)
// ---------------------------------------------------------------------------
__global__ __launch_bounds__(256) void glob_kernel(
    const float* __restrict__ lg, const float* __restrict__ fw,
    const float* __restrict__ fb, const float* __restrict__ fa,
    float* __restrict__ glob) {
  const int idx = blockIdx.x * 256 + threadIdx.x;
  if (idx >= Bb * Kk * Cc) return;
  const int c = idx & 511, bk = idx >> 9;
  const int k = bk % Kk, b = bk / Kk;
  float s = 0.f;
#pragma unroll
  for (int n = 0; n < 16; ++n)
    s += fw[n] * lg[(((size_t)(b * Nn + n)) * Kk + k) * Cc + c];
  s += fb[0];
  glob[((size_t)(b * Kk + k)) * Cc + c] = prelu_f(s, fa[0]);
}

// ---------------------------------------------------------------------------
// Kernel 8: v[b,k,d] = glob[b,k,:] . v_w[d,:] + v_b[d]   (tiny)
// ---------------------------------------------------------------------------
__global__ __launch_bounds__(256) void vproj_kernel(
    const float* __restrict__ glob, const float* __restrict__ vw,
    const float* __restrict__ vb, float* __restrict__ vv) {
  const int idx = blockIdx.x * 256 + threadIdx.x;
  if (idx >= Bb * Kk * CI) return;
  const int d = idx & 255, bk = idx >> 8;
  const float* gr = glob + (size_t)bk * Cc;
  const float* wr = vw + (size_t)d * Cc;
  float s = vb[d];
  for (int c = 0; c < Cc; ++c) s += gr[c] * wr[c];
  vv[idx] = s;
}

// ---------------------------------------------------------------------------
// Kernel 9 (fused): per (b, n, 16-pixel tile):
//   q = x_tile @ q_w^T + q_b  (WMMA K=512) -> logits = q @ kk^T (WMMA K=256)
//   -> softmax -> o = aff @ v (WMMA K=32) -> y = o @ out_conv_w^T (WMMA K=256)
//   -> BN -> PReLU -> residual.  grid (64,16,8), block 256. LDS = 51KB.
// ---------------------------------------------------------------------------
__global__ __launch_bounds__(256) void attn_out_kernel(
    const float* __restrict__ x,  const float* __restrict__ qw,
    const float* __restrict__ qb, const float* __restrict__ kkb,
    const float* __restrict__ vvb, const float* __restrict__ ow,
    const float* __restrict__ gamma, const float* __restrict__ beta,
    const float* __restrict__ mean,  const float* __restrict__ var,
    const float* __restrict__ pa,    float* __restrict__ out) {
  __shared__ _Float16 Abuf[16 * 512];   // x tile -> q (16x256) -> o (16x256)
  __shared__ _Float16 Bbuf[32 * BTILE]; // staged B tiles (qw / kk^T / v / ow)
  __shared__ float    logits[16 * 32];
  __shared__ _Float16 affA[16 * 32];

  const int pt = blockIdx.x, n = blockIdx.y, b = blockIdx.z;
  const int bh = n >> 2, bw = n & 3;
  const int p0 = pt * 16;
  const int hwbase = (bh * 32 + (p0 >> 5)) * Ww + bw * 32 + (p0 & 31);
  const int t = threadIdx.x, wv = t >> 5, lane = t & 31;
  const size_t bn = (size_t)(b * Nn + n);

  // 1. x tile -> Abuf [16 px x 512 ch], row-major; float4 over pixels
  for (int e = t; e < 512 * 4; e += 256) {
    int c = e >> 2, pix4 = (e & 3) * 4;
    f32x4 vv = *(const f32x4*)(x + ((size_t)(b * Cc + c)) * HW + hwbase + pix4);
#pragma unroll
    for (int i = 0; i < 4; ++i)
      Abuf[(pix4 + i) * 512 + c] = (_Float16)vv[i];
  }
  __syncthreads();

  // 2. q = x @ qw^T  (16 x 256), 2 N-tiles per wave
  v8f qacc[2] = {};
  for (int kc = 0; kc < 16; ++kc) {
    __builtin_prefetch(qw + ((size_t)((kc + 1) & 15)) * 32, 0, 0);
    for (int e = t; e < 256 * 8; e += 256) {      // float4 weight stage
      int d = e >> 3, cc4 = (e & 7) * 4;
      f32x4 vv = *(const f32x4*)(qw + ((size_t)d) * Cc + kc * 32 + cc4);
      *(v4h*)(Bbuf + BIDX(d, cc4)) = cvt4(vv);
    }
    __syncthreads();
    v16h af = load_a_frag(Abuf + kc * 32, 512, lane);
#pragma unroll
    for (int j = 0; j < 2; ++j) {
      v16h bf = load_b_frag(Bbuf + (wv * 2 + j) * BTILE, lane);
      qacc[j] = wmma_f16(af, bf, qacc[j]);
    }
    __syncthreads();
  }
  // write q (with bias) as f16 A-matrix into Abuf[16x256]
  {
    const int nn = lane & 15, mb = (lane >> 4) * 8;
#pragma unroll
    for (int j = 0; j < 2; ++j) {
      int d = (wv * 2 + j) * 16 + nn;
      float bd = qb[d];
#pragma unroll
      for (int i = 0; i < 8; ++i)
        Abuf[(mb + i) * 256 + d] = (_Float16)(qacc[j][i] + bd);
    }
  }
  // 3. stage kk^T (K=256 x N=32) as 8 k-chunks x 2 tiles, compute logits
  for (int e = t; e < 32 * 64; e += 256) {        // float4 over feature dim
    int kcol = e >> 6, cc4 = (e & 63) * 4;
    v4h val = {};
    if (kcol < Kk)
      val = cvt4(*(const f32x4*)(kkb + (bn * Kk + kcol) * CI + cc4));
    int base = ((cc4 >> 5) * 2 + (kcol >> 4)) * BTILE;
    *(v4h*)(Bbuf + base + (kcol & 15) * 32 + (cc4 & 31)) = val;
  }
  __syncthreads();
  if (wv < 2) {
    v8f lac = {};
#pragma unroll
    for (int kc = 0; kc < 8; ++kc) {
      v16h af = load_a_frag(Abuf + kc * 32, 256, lane);
      v16h bf = load_b_frag(Bbuf + (kc * 2 + wv) * BTILE, lane);
      lac = wmma_f16(af, bf, lac);
    }
    const int nn = lane & 15, mb = (lane >> 4) * 8;
#pragma unroll
    for (int i = 0; i < 8; ++i)
      logits[(mb + i) * 32 + wv * 16 + nn] = lac[i];
  }
  __syncthreads();

  // 4. softmax over k (19 valid) per pixel row
  if (t < 16) {
    float mx = -3.4e38f;
    for (int k = 0; k < Kk; ++k) mx = fmaxf(mx, logits[t * 32 + k]);
    float s = 0.f;
    for (int k = 0; k < Kk; ++k) {
      float ev = expf(logits[t * 32 + k] - mx);
      logits[t * 32 + k] = ev;
      s += ev;
    }
    float inv = 1.f / s;
    for (int k = 0; k < 32; ++k)
      affA[t * 32 + k] = (k < Kk) ? (_Float16)(logits[t * 32 + k] * inv)
                                  : (_Float16)0.f;
  }
  __syncthreads();

  // 5. o = aff @ v  (16 x 256, K=32 padded)
  for (int e = t; e < 32 * 64; e += 256) {        // float4 over d
    int kr = e >> 6, d4 = (e & 63) * 4;
    f32x4 vv = {};
    if (kr < Kk) vv = *(const f32x4*)(vvb + ((size_t)(b * Kk + kr)) * CI + d4);
#pragma unroll
    for (int i = 0; i < 4; ++i)
      Bbuf[BIDX(d4 + i, kr)] = (_Float16)vv[i];
  }
  __syncthreads();
  v8f oacc[2] = {};
  {
    v16h af = load_a_frag(affA, 32, lane);
#pragma unroll
    for (int j = 0; j < 2; ++j) {
      v16h bf = load_b_frag(Bbuf + (wv * 2 + j) * BTILE, lane);
      oacc[j] = wmma_f16(af, bf, oacc[j]);
    }
  }
  __syncthreads();
  {
    const int nn = lane & 15, mb = (lane >> 4) * 8;
#pragma unroll
    for (int j = 0; j < 2; ++j) {
      int d = (wv * 2 + j) * 16 + nn;
#pragma unroll
      for (int i = 0; i < 8; ++i)
        Abuf[(mb + i) * 256 + d] = (_Float16)oacc[j][i];
    }
  }
  __syncthreads();

  // 6. y = o @ out_conv_w^T  (16 x 512), 4 N-tiles per wave, K=256
  v8f yacc[4] = {};
  for (int kc = 0; kc < 8; ++kc) {
    __builtin_prefetch(ow + ((size_t)((kc + 1) & 7)) * 32, 0, 0);
    for (int e = t; e < 512 * 8; e += 256) {      // float4 weight stage
      int oc = e >> 3, cc4 = (e & 7) * 4;
      f32x4 vv = *(const f32x4*)(ow + ((size_t)oc) * CI + kc * 32 + cc4);
      *(v4h*)(Bbuf + BIDX(oc, cc4)) = cvt4(vv);
    }
    __syncthreads();
    v16h af = load_a_frag(Abuf + kc * 32, 256, lane);
#pragma unroll
    for (int j = 0; j < 4; ++j) {
      v16h bf = load_b_frag(Bbuf + (wv * 4 + j) * BTILE, lane);
      yacc[j] = wmma_f16(af, bf, yacc[j]);
    }
    __syncthreads();
  }

  // 7. BN + PReLU + residual, store
  const int nn = lane & 15, mb = (lane >> 4) * 8;
#pragma unroll
  for (int j = 0; j < 4; ++j) {
    int oc = (wv * 4 + j) * 16 + nn;
    float gm = gamma[oc], bt = beta[oc], mn = mean[oc];
    float iv = rsqrtf(var[oc] + 1e-5f);
    float a  = pa[oc];
#pragma unroll
    for (int i = 0; i < 8; ++i) {
      int pix = mb + i;
      float y = gm * (yacc[j][i] - mn) * iv + bt;
      y = prelu_f(y, a);
      size_t idx = ((size_t)(b * Cc + oc)) * HW + hwbase + pix;
      out[idx] = x[idx] + y;
    }
  }
}

// ---------------------------------------------------------------------------
// Host launcher
// ---------------------------------------------------------------------------
extern "C" void kernel_launch(void* const* d_in, const int* in_sizes, int n_in,
                              void* d_out, int out_size, void* d_ws, size_t ws_size,
                              hipStream_t stream) {
  (void)in_sizes; (void)n_in; (void)out_size; (void)ws_size;
  const float* x     = (const float*)d_in[0];
  const float* camw  = (const float*)d_in[1];
  const float* camb  = (const float*)d_in[2];
  const float* gcnw  = (const float*)d_in[3];
  const float* gcna  = (const float*)d_in[4];
  const float* linw  = (const float*)d_in[5];
  const float* fw    = (const float*)d_in[6];
  const float* fb    = (const float*)d_in[7];
  const float* fa    = (const float*)d_in[8];
  const float* qw    = (const float*)d_in[9];
  const float* qb    = (const float*)d_in[10];
  const float* kw    = (const float*)d_in[11];
  const float* kb    = (const float*)d_in[12];
  const float* vw    = (const float*)d_in[13];
  const float* vb    = (const float*)d_in[14];
  const float* ow    = (const float*)d_in[15];
  const float* gamma = (const float*)d_in[16];
  const float* beta  = (const float*)d_in[17];
  const float* mean  = (const float*)d_in[18];
  const float* var   = (const float*)d_in[19];
  const float* opa   = (const float*)d_in[20];
  float* out = (float*)d_out;

  // workspace layout (floats)
  float* ws   = (float*)d_ws;
  float* cam  = ws;                               // B*K*HW        = 2,490,368
  float* cls  = cam  + (size_t)Bb * Kk * HW;      // B*K*N         = 2,432
  float* conf = cls  + (size_t)Bb * Kk * Nn;      // B*N*K*P       = 2,490,368
  float* loc  = conf + (size_t)Bb * Nn * Kk * Pp; // B*N*K*C       = 1,245,184
  float* g    = loc  + (size_t)Bb * Nn * Kk * Cc; // B*N*K*C
  float* lg   = g    + (size_t)Bb * Nn * Kk * Cc; // B*N*K*C
  float* glob = lg   + (size_t)Bb * Nn * Kk * Cc; // B*K*C         = 77,824
  float* kkb  = glob + (size_t)Bb * Kk * Cc;      // B*N*K*CI      = 622,592
  float* vvb  = kkb  + (size_t)Bb * Nn * Kk * CI; // B*K*CI        = 38,912

  cam_kernel<<<dim3(HW / 256, Bb), 256, 0, stream>>>(x, camw, camb, cam);
  binstat_kernel<<<dim3(Nn, Kk, Bb), 256, 0, stream>>>(cam, cls, conf);
  local_kernel<<<Bb * Nn, 256, 0, stream>>>(x, conf, cls, loc);
  gcn_kernel<<<(Bb * Kk * Cc + 255) / 256, 256, 0, stream>>>(loc, gcnw, gcna, g);
  rowgemm_kernel<<<dim3(4, 152), 256, 0, stream>>>(g, linw, nullptr, lg, Cc);
  glob_kernel<<<(Bb * Kk * Cc + 255) / 256, 256, 0, stream>>>(lg, fw, fb, fa, glob);
  rowgemm_kernel<<<dim3(2, 152), 256, 0, stream>>>(lg, kw, kb, kkb, CI);
  vproj_kernel<<<(Bb * Kk * CI + 255) / 256, 256, 0, stream>>>(glob, vw, vb, vvb);
  attn_out_kernel<<<dim3(Pp / 16, Nn, Bb), 256, 0, stream>>>(
      x, qw, qb, kkb, vvb, ow, gamma, beta, mean, var, opa, out);
}
